// ResidualAttention_39711267619247
// MI455X (gfx1250) — matrix operand order repaired and checked by reading, not compiled, not run on hardware
//
#include <hip/hip_runtime.h>
#include <hip/hip_bf16.h>

// ---------------------------------------------------------------------------
// Residual attention block (B=2, S=2048, H=1024, 16 heads x 64) for gfx1250.
// Matmuls on v_wmma_f32_16x16x32_f16; f16 tile staging via TENSOR_LOAD_TO_LDS.
// ---------------------------------------------------------------------------

#define HIDDEN   1024
#define HEADS    16
#define HEAD_DIM 64
#define SEQ      2048
#define BATCH    2
#define ROWS     (BATCH * SEQ)   // 4096

typedef __attribute__((ext_vector_type(16))) _Float16 v16h;
typedef __attribute__((ext_vector_type(8)))  _Float16 v8h;
typedef __attribute__((ext_vector_type(4)))  _Float16 v4h;
typedef __attribute__((ext_vector_type(8)))  float    v8f;
typedef __attribute__((ext_vector_type(4)))  float    v4f;
typedef __attribute__((ext_vector_type(4)))  unsigned int v4u;
typedef __attribute__((ext_vector_type(8)))  int      v8i;
typedef __attribute__((ext_vector_type(4)))  int      v4i;

__device__ __forceinline__ v8f zero_v8f() {
    v8f z;
#pragma unroll
    for (int i = 0; i < 8; ++i) z[i] = 0.0f;
    return z;
}

// D = A*B + C, 16x16x32 f16 -> f32
__device__ __forceinline__ v8f wmma_f16(v16h a, v16h b, v8f c) {
    return __builtin_amdgcn_wmma_f32_16x16x32_f16(false, a, false, b,
                                                  (short)0, c, false, false);
}

// A-fragment (16x32 f16, MxK). Lane l<16 : row M=l, K=0..7 & 16..23.
// Lane l>=16: row M=l-16, K=8..15 & 24..31. LDS tile is row-major [M][ldk].
__device__ __forceinline__ v16h load_a_frag(const _Float16* base, int lane, int ldk) {
    const int r  = lane & 15;
    const int hi = lane >> 4;
    const _Float16* p = base + r * ldk + hi * 8;
    v8h lo = *(const v8h*)(p);
    v8h up = *(const v8h*)(p + 16);
    v16h f;
#pragma unroll
    for (int i = 0; i < 8; ++i) { f[i] = lo[i]; f[i + 8] = up[i]; }
    return f;
}

// B-fragment (32x16 f16, KxN). Lane l<16: col N=l, K=0..7 & 16..23; lane>=16: K=8..15 & 24..31.
// LDS tile stored K-transposed: [N][ldk] with k contiguous -> two v8h reads.
__device__ __forceinline__ v16h load_b_frag(const _Float16* base, int lane, int ldk) {
    const int n  = lane & 15;
    const int hi = lane >> 4;
    const _Float16* p = base + n * ldk + hi * 8;
    v8h lo = *(const v8h*)(p);
    v8h up = *(const v8h*)(p + 16);
    v16h f;
#pragma unroll
    for (int i = 0; i < 8; ++i) { f[i] = lo[i]; f[i + 8] = up[i]; }
    return f;
}

// Low 32 bits of a generic pointer into LDS = workgroup-relative LDS address.
__device__ __forceinline__ unsigned lds_addr_of(const void* p) {
    return (unsigned)(unsigned long long)(uintptr_t)p;
}

// ---------------------------------------------------------------------------
// TDM: 2-D tile load Global -> LDS, 16-bit elements.
//   tile_d0 : contiguous elements per row      tile_d1 : rows
//   stride0 : global stride between rows (elements)
// Descriptor layout per CDNA5 ISA 8.3/8.4 (count=1, type=2, data_size=2B).
// Issue from ONE wave; complete with s_wait_tensorcnt.
// ---------------------------------------------------------------------------
__device__ __forceinline__ void tdm_load_2d_f16(unsigned lds_dst, const void* gsrc,
                                                unsigned tile_d0, unsigned tile_d1,
                                                unsigned long long stride0,
                                                unsigned tensor_d0, unsigned tensor_d1) {
    const unsigned long long ga = (unsigned long long)(uintptr_t)gsrc;
    v4u g0;
    g0[0] = 1u;                                         // count=1, user mode
    g0[1] = lds_dst;                                    // lds_addr (bytes)
    g0[2] = (unsigned)ga;                               // global_addr[31:0]
    g0[3] = (unsigned)((ga >> 32) & 0x1FFFFFFu)         // global_addr[56:32]
          | (2u << 30);                                 // type = 2 ("image")
    v8i g1;
    g1[0] = (int)(1u << 16);                            // wg_mask=0, data_size=2B
    g1[1] = (int)((tensor_d0 & 0xFFFFu) << 16);         // tensor_dim0[15:0]
    g1[2] = (int)(((tensor_d0 >> 16) & 0xFFFFu)
          | ((tensor_d1 & 0xFFFFu) << 16));             // dim0 hi | dim1 lo
    g1[3] = (int)(((tensor_d1 >> 16) & 0xFFFFu)
          | ((tile_d0 & 0xFFFFu) << 16));               // dim1 hi | tile_dim0
    g1[4] = (int)(tile_d1 & 0xFFFFu);                   // tile_dim1, tile_dim2=0
    g1[5] = (int)(unsigned)(stride0 & 0xFFFFFFFFu);     // dim0_stride[31:0]
    g1[6] = (int)(unsigned)((stride0 >> 32) & 0xFFFFu); // dim0_stride[47:32]
    g1[7] = 0;                                          // dim1_stride unused (2D)
    v4i gz4;
    gz4[0] = 0; gz4[1] = 0; gz4[2] = 0; gz4[3] = 0;
    v8i gz8;
#pragma unroll
    for (int i = 0; i < 8; ++i) gz8[i] = 0;
    // 6-arg form (clang-23 / therock-10.0 signature)
    __builtin_amdgcn_tensor_load_to_lds(g0, g1, gz4, gz4, gz8, 0);
}

// ---------------------------------------------------------------------------
// Kernel 1: fused QKV projection.  X[4096,1024] (f32) x W[1024,1024] (f32)
// -> f16 outputs.  z=0: Q (scaled by 1/8) [b][h][s][d]
//                  z=1: K               [b][h][s][d]
//                  z=2: V transposed    [b][h][d][s]
// Block: 256 thr (8 waves, 4x2 wave grid), tile 128x128, BK=32.
// (f32 sources must be converted, so tiles stage through VALU, not TDM.)
// ---------------------------------------------------------------------------
__global__ void __launch_bounds__(256)
qkv_gemm_kernel(const float* __restrict__ X,
                const float* __restrict__ Wq, const float* __restrict__ bq,
                const float* __restrict__ Wk, const float* __restrict__ bk,
                const float* __restrict__ Wv, const float* __restrict__ bv,
                _Float16* __restrict__ qf, _Float16* __restrict__ kf,
                _Float16* __restrict__ vt) {
    __shared__ _Float16 As[128 * 32];   // [m][k]
    __shared__ _Float16 Bt[128 * 32];   // [n][k]  (W transposed)

    const int t    = threadIdx.x;
    const int lane = t & 31;
    const int wave = t >> 5;
    const int wm   = wave >> 1;     // 0..3 -> M offset wm*32
    const int wn   = wave & 1;      // 0..1 -> N offset wn*64
    const int nBase = blockIdx.x * 128;
    const int mBase = blockIdx.y * 128;
    const int z     = blockIdx.z;

    const float* W    = (z == 0) ? Wq : (z == 1) ? Wk : Wv;
    const float* bias = (z == 0) ? bq : (z == 1) ? bk : bv;

    v8f acc[2][4];
#pragma unroll
    for (int mi = 0; mi < 2; ++mi)
#pragma unroll
        for (int ni = 0; ni < 4; ++ni) acc[mi][ni] = zero_v8f();

    for (int k0 = 0; k0 < HIDDEN; k0 += 32) {
        if (k0 + 32 < HIDDEN) {   // warm the next tiles (global_prefetch_b8)
            __builtin_prefetch(X + (size_t)(mBase + (t >> 1)) * HIDDEN + k0 + 32, 0, 1);
            __builtin_prefetch(W + (size_t)(k0 + 32 + (t >> 3)) * HIDDEN + nBase, 0, 1);
        }
        // A tile: 128x32 f32 -> f16
#pragma unroll
        for (int i = 0; i < 4; ++i) {
            int idx4 = t + i * 256;                 // float4 index, 0..1023
            int r = idx4 >> 3, c = (idx4 & 7) << 2;
            v4f x = *(const v4f*)(X + (size_t)(mBase + r) * HIDDEN + k0 + c);
            v4h h;
            h[0] = (_Float16)x[0]; h[1] = (_Float16)x[1];
            h[2] = (_Float16)x[2]; h[3] = (_Float16)x[3];
            *(v4h*)(As + r * 32 + c) = h;
        }
        // B tile: W[k0..k0+32][nBase..+128] -> Bt[n][k] f16
#pragma unroll
        for (int i = 0; i < 4; ++i) {
            int idx4 = t + i * 256;
            int kk = idx4 >> 5, c = (idx4 & 31) << 2;
            v4f w = *(const v4f*)(W + (size_t)(k0 + kk) * HIDDEN + nBase + c);
            Bt[(c + 0) * 32 + kk] = (_Float16)w[0];
            Bt[(c + 1) * 32 + kk] = (_Float16)w[1];
            Bt[(c + 2) * 32 + kk] = (_Float16)w[2];
            Bt[(c + 3) * 32 + kk] = (_Float16)w[3];
        }
        __syncthreads();

        v16h af[2], bf[4];
#pragma unroll
        for (int mi = 0; mi < 2; ++mi)
            af[mi] = load_a_frag(As + (wm * 32 + mi * 16) * 32, lane, 32);
#pragma unroll
        for (int ni = 0; ni < 4; ++ni)
            bf[ni] = load_b_frag(Bt + (wn * 64 + ni * 16) * 32, lane, 32);
#pragma unroll
        for (int mi = 0; mi < 2; ++mi)
#pragma unroll
            for (int ni = 0; ni < 4; ++ni)
                acc[mi][ni] = wmma_f16(af[mi], bf[ni], acc[mi][ni]);
        __syncthreads();
    }

    // Epilogue: +bias, optional 1/sqrt(head_dim) fold into Q, layout shuffle.
    const int hi = lane >> 4, ln = lane & 15;
    const float qscale = (z == 0) ? 0.125f : 1.0f;
    _Float16* outp = (z == 0) ? qf : (z == 1) ? kf : vt;
#pragma unroll
    for (int ni = 0; ni < 4; ++ni) {
        const int n  = nBase + wn * 64 + ni * 16 + ln;
        const float bn = bias[n];
        const int hh = n >> 6, dd = n & 63;
#pragma unroll
        for (int mi = 0; mi < 2; ++mi) {
#pragma unroll
            for (int j = 0; j < 8; ++j) {
                const int m = mBase + wm * 32 + mi * 16 + j + hi * 8;
                const int b = m >> 11, s = m & (SEQ - 1);
                const float v = (acc[mi][ni][j] + bn) * qscale;
                size_t idx;
                if (z < 2)
                    idx = (((size_t)b * HEADS + hh) * SEQ + s) * HEAD_DIM + dd;
                else
                    idx = (((size_t)b * HEADS + hh) * HEAD_DIM + dd) * SEQ + s;
                outp[idx] = (_Float16)v;
            }
        }
    }
}

// ---------------------------------------------------------------------------
// Kernel 2: flash attention.  One block = 128 queries of one (b,h).
// Q block + 64-key K / V^T chunks DMA'd into LDS by the Tensor Data Mover;
// online softmax in registers; both matmuls on WMMA.
// ---------------------------------------------------------------------------
__global__ void __launch_bounds__(256)
attention_kernel(const _Float16* __restrict__ qf, const _Float16* __restrict__ kf,
                 const _Float16* __restrict__ vt, _Float16* __restrict__ ctx) {
    __shared__ _Float16 Qs[128 * 64];   // [q][d]
    __shared__ _Float16 Ks[64 * 64];    // [k][d]
    __shared__ _Float16 Vs[64 * 64];    // [d][k]  (V already transposed)
    __shared__ _Float16 Ps[128 * 64];   // probs f16, wave-private rows

    const int t    = threadIdx.x;
    const int lane = t & 31;
    const int wave = t >> 5;            // 0..7 -> query rows wave*16..+15
    const int hi   = lane >> 4, ln = lane & 15;
    const int qb   = blockIdx.x;        // 0..15
    const int bh   = blockIdx.y;        // 0..31

    const size_t baseQK = (size_t)bh * SEQ * HEAD_DIM;
    const _Float16* qp = qf + baseQK + (size_t)qb * 128 * HEAD_DIM;
    const _Float16* kp = kf + baseQK;
    const _Float16* vp = vt + baseQK;   // [d][s]

    // Kick the Q block DMA immediately (TDM runs while we init registers).
    if (wave == 0)
        tdm_load_2d_f16(lds_addr_of(Qs), qp, /*tile*/64, 128,
                        /*stride0*/64, /*tensor*/64, 128);

    float m_i[8], l_i[8];
#pragma unroll
    for (int j = 0; j < 8; ++j) { m_i[j] = -1e30f; l_i[j] = 0.0f; }
    v8f o[4];
#pragma unroll
    for (int tt = 0; tt < 4; ++tt) o[tt] = zero_v8f();
    v16h aq[2];   // Q fragments, loaded once after the DMA lands

    for (int kv = 0; kv < SEQ; kv += 64) {
        __syncthreads();   // previous chunk fully consumed
        if (wave == 0) {
            tdm_load_2d_f16(lds_addr_of(Ks), kp + (size_t)kv * HEAD_DIM,
                            /*tile*/64, 64, /*stride0*/64, 64, 64);
            tdm_load_2d_f16(lds_addr_of(Vs), vp + kv,
                            /*tile*/64, 64, /*stride0*/SEQ, 64, 64);
            __builtin_amdgcn_s_wait_tensorcnt(0);   // Q on first pass, K/V each pass
        }
        __syncthreads();

        if (kv == 0) {   // hoist invariant Q fragments into registers
#pragma unroll
            for (int ks = 0; ks < 2; ++ks)
                aq[ks] = load_a_frag(Qs + (wave * 16) * 64 + ks * 32, lane, 64);
        }

        // ---- scores S = Q K^T (scale pre-folded into Q) ----
        v8f sc[4];
#pragma unroll
        for (int ni = 0; ni < 4; ++ni) sc[ni] = zero_v8f();
#pragma unroll
        for (int ks = 0; ks < 2; ++ks) {
#pragma unroll
            for (int ni = 0; ni < 4; ++ni) {
                v16h bk = load_b_frag(Ks + (ni * 16) * 64 + ks * 32, lane, 64);
                sc[ni] = wmma_f16(aq[ks], bk, sc[ni]);
            }
        }

        // ---- online softmax (row = wave*16 + j + 8*hi, spread over 16 lanes)
#pragma unroll
        for (int j = 0; j < 8; ++j) {
            float mx = sc[0][j];
#pragma unroll
            for (int ni = 1; ni < 4; ++ni) mx = fmaxf(mx, sc[ni][j]);
#pragma unroll
            for (int off = 8; off >= 1; off >>= 1)
                mx = fmaxf(mx, __shfl_xor(mx, off, 32));
            const float m_new = fmaxf(m_i[j], mx);
            const float scale = __expf(m_i[j] - m_new);
            float rs = 0.0f;
#pragma unroll
            for (int ni = 0; ni < 4; ++ni) {
                float p = __expf(sc[ni][j] - m_new);
                sc[ni][j] = p;
                rs += p;
            }
#pragma unroll
            for (int off = 8; off >= 1; off >>= 1)
                rs += __shfl_xor(rs, off, 32);
            l_i[j] = l_i[j] * scale + rs;
            m_i[j] = m_new;
#pragma unroll
            for (int tt = 0; tt < 4; ++tt) o[tt][j] *= scale;
        }

        // ---- probs f32 (C layout) -> f16 A layout via wave-private LDS ----
#pragma unroll
        for (int ni = 0; ni < 4; ++ni)
#pragma unroll
            for (int j = 0; j < 8; ++j)
                Ps[(wave * 16 + j + hi * 8) * 64 + ni * 16 + ln] =
                    (_Float16)sc[ni][j];
        asm volatile("s_wait_dscnt 0" ::: "memory");   // same-wave LDS RAW

        // ---- O += P V ----
#pragma unroll
        for (int ks = 0; ks < 2; ++ks) {
            v16h ap = load_a_frag(Ps + (wave * 16) * 64 + ks * 32, lane, 64);
#pragma unroll
            for (int tt = 0; tt < 4; ++tt) {
                v16h bv = load_b_frag(Vs + (tt * 16) * 64 + ks * 32, lane, 64);
                o[tt] = wmma_f16(ap, bv, o[tt]);
            }
        }
    }

    // ---- normalize and store context as [b][s][H] f16 ----
    const int b = bh >> 4, h = bh & 15;
#pragma unroll
    for (int j = 0; j < 8; ++j) {
        const float inv_l = 1.0f / l_i[j];
        const int row = qb * 128 + wave * 16 + j + hi * 8;
#pragma unroll
        for (int tt = 0; tt < 4; ++tt) {
            ctx[((size_t)b * SEQ + row) * HIDDEN + h * HEAD_DIM + tt * 16 + ln] =
                (_Float16)(o[tt][j] * inv_l);
        }
    }
}

// ---------------------------------------------------------------------------
// Kernel 3: output projection + bias + residual -> f32 (pre-layernorm).
// A-tile (f16 context) is DMA'd by the TDM while the VALU converts/transposes
// the f32 W tile -- DMA and conversion overlap every K-step.
// ---------------------------------------------------------------------------
__global__ void __launch_bounds__(256)
oproj_kernel(const _Float16* __restrict__ ctx, const float* __restrict__ Wo,
             const float* __restrict__ bo, const float* __restrict__ resid,
             float* __restrict__ out) {
    __shared__ _Float16 As[128 * 32];
    __shared__ _Float16 Bt[128 * 32];

    const int t    = threadIdx.x;
    const int lane = t & 31;
    const int wave = t >> 5;
    const int wm   = wave >> 1;
    const int wn   = wave & 1;
    const int nBase = blockIdx.x * 128;
    const int mBase = blockIdx.y * 128;

    v8f acc[2][4];
#pragma unroll
    for (int mi = 0; mi < 2; ++mi)
#pragma unroll
        for (int ni = 0; ni < 4; ++ni) acc[mi][ni] = zero_v8f();

    for (int k0 = 0; k0 < HIDDEN; k0 += 32) {
        // A tile via TDM: 128 rows x 32 cols, row stride HIDDEN (f16)
        if (wave == 0)
            tdm_load_2d_f16(lds_addr_of(As),
                            ctx + (size_t)mBase * HIDDEN + k0,
                            /*tile*/32, 128, /*stride0*/HIDDEN, 32, 128);
        if (k0 + 32 < HIDDEN)
            __builtin_prefetch(Wo + (size_t)(k0 + 32 + (t >> 3)) * HIDDEN + nBase, 0, 1);
        // B tile (f32 -> f16, transposed) -- overlaps with the TDM transfer
#pragma unroll
        for (int i = 0; i < 4; ++i) {
            int idx4 = t + i * 256;
            int kk = idx4 >> 5, c = (idx4 & 31) << 2;
            v4f w = *(const v4f*)(Wo + (size_t)(k0 + kk) * HIDDEN + nBase + c);
            Bt[(c + 0) * 32 + kk] = (_Float16)w[0];
            Bt[(c + 1) * 32 + kk] = (_Float16)w[1];
            Bt[(c + 2) * 32 + kk] = (_Float16)w[2];
            Bt[(c + 3) * 32 + kk] = (_Float16)w[3];
        }
        if (wave == 0) __builtin_amdgcn_s_wait_tensorcnt(0);
        __syncthreads();

        v16h af[2], bf[4];
#pragma unroll
        for (int mi = 0; mi < 2; ++mi)
            af[mi] = load_a_frag(As + (wm * 32 + mi * 16) * 32, lane, 32);
#pragma unroll
        for (int ni = 0; ni < 4; ++ni)
            bf[ni] = load_b_frag(Bt + (wn * 64 + ni * 16) * 32, lane, 32);
#pragma unroll
        for (int mi = 0; mi < 2; ++mi)
#pragma unroll
            for (int ni = 0; ni < 4; ++ni)
                acc[mi][ni] = wmma_f16(af[mi], bf[ni], acc[mi][ni]);
        __syncthreads();
    }

    const int hi = lane >> 4, ln = lane & 15;
#pragma unroll
    for (int ni = 0; ni < 4; ++ni) {
        const int n = nBase + wn * 64 + ni * 16 + ln;
        const float bn = bo[n];
#pragma unroll
        for (int mi = 0; mi < 2; ++mi) {
#pragma unroll
            for (int j = 0; j < 8; ++j) {
                const int m = mBase + wm * 32 + mi * 16 + j + hi * 8;
                const size_t idx = (size_t)m * HIDDEN + n;
                out[idx] = acc[mi][ni][j] + bn + resid[idx];
            }
        }
    }
}

// ---------------------------------------------------------------------------
// Kernel 4: in-place row layernorm over H=1024.  One block per row.
// ---------------------------------------------------------------------------
__global__ void __launch_bounds__(256)
layernorm_kernel(float* __restrict__ x, const float* __restrict__ gamma,
                 const float* __restrict__ beta) {
    __shared__ float red[16];
    const int t = threadIdx.x;
    float* xr = x + (size_t)blockIdx.x * HIDDEN;

    v4f v = *(const v4f*)(xr + t * 4);
    float s  = v[0] + v[1] + v[2] + v[3];
    float s2 = v[0] * v[0] + v[1] * v[1] + v[2] * v[2] + v[3] * v[3];
#pragma unroll
    for (int off = 16; off >= 1; off >>= 1) {
        s  += __shfl_xor(s,  off, 32);
        s2 += __shfl_xor(s2, off, 32);
    }
    const int wave = t >> 5, lane = t & 31;
    if (lane == 0) { red[wave] = s; red[8 + wave] = s2; }
    __syncthreads();
    float ts = 0.0f, ts2 = 0.0f;
#pragma unroll
    for (int i = 0; i < 8; ++i) { ts += red[i]; ts2 += red[8 + i]; }
    const float mean = ts * (1.0f / HIDDEN);
    const float var  = ts2 * (1.0f / HIDDEN) - mean * mean;
    const float rstd = rsqrtf(var + 1e-5f);

    v4f g = *(const v4f*)(gamma + t * 4);
    v4f b = *(const v4f*)(beta + t * 4);
    v4f y;
#pragma unroll
    for (int i = 0; i < 4; ++i) y[i] = (v[i] - mean) * rstd * g[i] + b[i];
    *(v4f*)(xr + t * 4) = y;
}

// ---------------------------------------------------------------------------
extern "C" void kernel_launch(void* const* d_in, const int* in_sizes, int n_in,
                              void* d_out, int out_size, void* d_ws, size_t ws_size,
                              hipStream_t stream) {
    (void)in_sizes; (void)n_in; (void)out_size; (void)ws_size;
    const float* X     = (const float*)d_in[0];
    const float* Wq    = (const float*)d_in[1];
    const float* bq    = (const float*)d_in[2];
    const float* Wk    = (const float*)d_in[3];
    const float* bk    = (const float*)d_in[4];
    const float* Wv    = (const float*)d_in[5];
    const float* bv    = (const float*)d_in[6];
    const float* Wo    = (const float*)d_in[7];
    const float* bo    = (const float*)d_in[8];
    const float* gamma = (const float*)d_in[9];
    const float* beta  = (const float*)d_in[10];
    float* out = (float*)d_out;

    char* ws = (char*)d_ws;
    const size_t MB8 = (size_t)8 * 1024 * 1024;
    _Float16* qf  = (_Float16*)(ws);            // [b][h][s][d]
    _Float16* kf  = (_Float16*)(ws + 1 * MB8);  // [b][h][s][d]
    _Float16* vt  = (_Float16*)(ws + 2 * MB8);  // [b][h][d][s]
    _Float16* ctx = (_Float16*)(ws + 3 * MB8);  // [b][s][H]

    dim3 g1(HIDDEN / 128, ROWS / 128, 3);
    qkv_gemm_kernel<<<g1, 256, 0, stream>>>(X, Wq, bq, Wk, bk, Wv, bv, qf, kf, vt);

    dim3 g2(SEQ / 128, BATCH * HEADS);
    attention_kernel<<<g2, 256, 0, stream>>>(qf, kf, vt, ctx);

    dim3 g3(HIDDEN / 128, ROWS / 128);
    oproj_kernel<<<g3, 256, 0, stream>>>(ctx, Wo, bo, X, out);

    layernorm_kernel<<<ROWS, 256, 0, stream>>>(out, gamma, beta);
}
